// Model0_31731218382992
// MI455X (gfx1250) — matrix operand order repaired and checked by reading, not compiled, not run on hardware
//
#include <hip/hip_runtime.h>

typedef __bf16 bf16;
typedef __attribute__((ext_vector_type(16))) __bf16 v16bf;
typedef __attribute__((ext_vector_type(8)))  float  v8f;

#define BT  16
#define KNB 7
#define CN5 10242
#define CN4 2562
#define CN3 642

// ---------------- elementwise f32 -> bf16 ----------------
__global__ void cvt_f32_bf16(const float* __restrict__ in, bf16* __restrict__ out, int n) {
    int i = blockIdx.x * blockDim.x + threadIdx.x;
    if (i < n) out[i] = (bf16)in[i];
}

// ---------------- pack W [K,CIN,COUT] f32 -> WMMA B fragments ----------------
// layout: dword index = ((ct*RK + rk)*32 + lane)*8 + v
// lane holds column co = ct*16 + (lane&15); kk = (lane>>4)*8 + (v&3)*2 + (v>>2)*16 + pair
__global__ void pack_w(const float* __restrict__ W, unsigned int* __restrict__ out,
                       int CIN, int COUT) {
    int RK = KNB * (CIN >> 5);
    int total = (COUT >> 4) * RK * 32 * 8;
    int i = blockIdx.x * blockDim.x + threadIdx.x;
    if (i >= total) return;
    int v    = i & 7;
    int lane = (i >> 3) & 31;
    int rk   = (i >> 8) % RK;
    int ct   = (i >> 8) / RK;
    int co   = ct * 16 + (lane & 15);
    int kk0  = (lane >> 4) * 8 + (v & 3) * 2 + (v >> 2) * 16;
    unsigned int res = 0;
    for (int p = 0; p < 2; ++p) {
        int r = rk * 32 + kk0 + p;
        int k = r / CIN, c = r % CIN;
        union { bf16 h; unsigned short s; } u;
        u.h = (bf16)W[((size_t)k * CIN + c) * COUT + co];
        res |= ((unsigned int)u.s) << (16 * p);
    }
    out[i] = res;
}

// ---------------- WMMA sphere-conv GEMM with LDS-staged weights ----------------
// out[bt,n,co] = relu( sum_k sum_c xin[bt, nbr[n,k], c] * W[k,c,co] + bias[co] )
// block = 8 waves sharing (bt, 64-col group); each wave = 32 vertices x 64 cols
// (2 row subtiles x 4 col tiles = 8 WMMA accumulators; B reused by both rows).
// B fragments for one neighbor-k slice are staged in LDS once per block.
template<int CIN, int COUT>
__global__ void __launch_bounds__(256)
conv_wmma(const bf16* __restrict__ xin,            // [BT, Nin, CIN]
          const int* __restrict__ nbr,             // [N, 7]
          const uint4* __restrict__ packB,         // packed fragments (16B units)
          const float* __restrict__ bias,          // [COUT]
          bf16* __restrict__ out,                  // [BT, N, COUT]
          int N, int Nin, int relu) {
    constexpr int cpk = CIN >> 5;          // 32-wide chunks per neighbor k
    constexpr int RK  = KNB * cpk;
    constexpr int cgN = COUT >> 6;         // 64-wide column groups
    constexpr int SL16 = 4 * cpk * 64;     // uint4 units per k-slice (4 tiles)
    __shared__ uint4 smem[SL16];           // <= 48KB (conv7)

    const int tiles32  = (N + 31) >> 5;    // 32-row tiles
    const int ntGroups = (tiles32 + 7) >> 3;
    const int tid  = threadIdx.x;
    const int lane = tid & 31;
    const int w    = tid >> 5;             // wave id in block (row tile select)

    // block decode: (bt, ntGroup, cg)
    int bid = blockIdx.x;
    int cg  = bid % cgN;  bid /= cgN;
    int ntg = bid % ntGroups;
    int bt  = bid / ntGroups;

    int nt    = ntg * 8 + w;
    int valid = nt < tiles32;
    if (!valid) nt = tiles32 - 1;          // clamp; stores masked below

    const int half = lane >> 4;
    const int col  = lane & 15;            // A row index == D column index
    const int vtx0 = nt << 5;
    int vtxA = vtx0 + col;      if (vtxA >= N) vtxA = N - 1;
    int vtxB = vtx0 + 16 + col; if (vtxB >= N) vtxB = N - 1;

    // per-neighbor gathered row base pointers (chunk offset folds into imm)
    const bf16* abaseA[KNB];
    const bf16* abaseB[KNB];
#pragma unroll
    for (int k = 0; k < KNB; ++k) {
        abaseA[k] = xin + ((size_t)bt * Nin + nbr[vtxA * KNB + k]) * CIN + half * 8;
        abaseB[k] = xin + ((size_t)bt * Nin + nbr[vtxB * KNB + k]) * CIN + half * 8;
    }

    // cooperative B staging: slice k of tile j lives at global uint4 index
    //   ((cg*4+j)*RK + k*cpk)*64 + rem,  rem = cc*64 + lane*2 + h
    // LDS layout mirrors it: smem[(j*cpk+cc)*64 + lane*2 + h]
    auto gidx = [&](int k, int u) -> size_t {
        int j   = u / (cpk * 64);
        int rem = u - j * (cpk * 64);
        return ((size_t)((cg * 4 + j) * RK + k * cpk)) * 64 + rem;
    };

    uint4 st[cpk];
#pragma unroll
    for (int i = 0; i < cpk; ++i) st[i] = packB[gidx(0, tid + i * 256)];
#pragma unroll
    for (int i = 0; i < cpk; ++i) smem[tid + i * 256] = st[i];
    __syncthreads();

    v8f acc[2][4];
#pragma unroll
    for (int r = 0; r < 2; ++r)
#pragma unroll
        for (int j = 0; j < 4; ++j) acc[r][j] = (v8f){};

    for (int k = 0; k < KNB; ++k) {
        if (k + 1 < KNB) {                 // issue next slice's global loads early
#pragma unroll
            for (int i = 0; i < cpk; ++i) st[i] = packB[gidx(k + 1, tid + i * 256)];
        }
        const bf16* aa = abaseA[k];
        const bf16* ab = abaseB[k];
#pragma unroll
        for (int cc = 0; cc < cpk; ++cc) {
            union { uint4 q[2]; v16bf v; } A0, A1;
            A0.q[0] = *reinterpret_cast<const uint4*>(aa + cc * 32);       // K=base+0..7
            A0.q[1] = *reinterpret_cast<const uint4*>(aa + cc * 32 + 16);  // K=base+16..23
            A1.q[0] = *reinterpret_cast<const uint4*>(ab + cc * 32);
            A1.q[1] = *reinterpret_cast<const uint4*>(ab + cc * 32 + 16);
#pragma unroll
            for (int j = 0; j < 4; ++j) {
                const uint4* sp = &smem[(j * cpk + cc) * 64 + lane * 2];
                union { uint4 q[2]; v16bf v; } Bf;
                Bf.q[0] = sp[0]; Bf.q[1] = sp[1];
                acc[0][j] = __builtin_amdgcn_wmma_f32_16x16x32_bf16(
                    false, A0.v, false, Bf.v, (short)0, acc[0][j], false, false);
                acc[1][j] = __builtin_amdgcn_wmma_f32_16x16x32_bf16(
                    false, A1.v, false, Bf.v, (short)0, acc[1][j], false, false);
            }
        }
        __syncthreads();                   // everyone done reading slice k
        if (k + 1 < KNB) {
#pragma unroll
            for (int i = 0; i < cpk; ++i) smem[tid + i * 256] = st[i];
            __syncthreads();               // slice k+1 published
        }
    }

#pragma unroll
    for (int r = 0; r < 2; ++r) {
#pragma unroll
        for (int j = 0; j < 4; ++j) {
            int co = (cg * 4 + j) * 16 + col;
            float bv = bias[co];
#pragma unroll
            for (int v = 0; v < 8; ++v) {
                int m = half * 8 + v;      // D: lanes 0-15 -> M=v, 16-31 -> M=8+v
                int n = vtx0 + r * 16 + m;
                if (valid && n < N) {
                    float val = acc[r][j][v] + bv;
                    if (relu && val < 0.f) val = 0.f;
                    out[((size_t)bt * N + n) * COUT + co] = (bf16)val;
                }
            }
        }
    }
}

// ---------------- max pool over 7 gathered fine vertices ----------------
__global__ void pool_max(const bf16* __restrict__ in, const int* __restrict__ idx,
                         bf16* __restrict__ out, int Nc, int Nf, int C) {
    int i = blockIdx.x * blockDim.x + threadIdx.x;
    if (i >= BT * Nc * C) return;
    int c  = i % C;
    int n  = (i / C) % Nc;
    int bt = i / (C * Nc);
    float m = -3.4e38f;
    for (int k = 0; k < KNB; ++k) {
        int s = idx[n * KNB + k];
        float v = (float)in[((size_t)bt * Nf + s) * C + c];
        m = v > m ? v : m;
    }
    out[i] = (bf16)m;
}

// ---------------- unpool-gather + channel concat ----------------
__global__ void concat_up(const bf16* __restrict__ up_src, const int* __restrict__ up_idx,
                          const bf16* __restrict__ same, bf16* __restrict__ out,
                          int Nf, int Nc, int C1, int C2) {
    int C = C1 + C2;
    size_t total = (size_t)BT * Nf * C;
    size_t i = (size_t)blockIdx.x * blockDim.x + threadIdx.x;
    if (i >= total) return;
    int c  = (int)(i % C);
    int n  = (int)((i / C) % Nf);
    int bt = (int)(i / ((size_t)C * Nf));
    bf16 v;
    if (c < C1) v = up_src[((size_t)bt * Nc + up_idx[n]) * C1 + c];
    else        v = same[((size_t)bt * Nf + n) * C2 + (c - C1)];
    out[i] = v;
}

// ---------------- 1-channel head (f32 accumulate, vectorized reads) ----------------
__global__ void conv9_k(const bf16* __restrict__ in, const int* __restrict__ nbr,
                        const float* __restrict__ W, const float* __restrict__ b,
                        float* __restrict__ logits, int N, int C) {
    int i = blockIdx.x * blockDim.x + threadIdx.x;
    if (i >= BT * N) return;
    int n = i % N, bt = i / N;
    float acc = b[0];
    for (int k = 0; k < KNB; ++k) {
        const bf16* p = in + ((size_t)bt * N + nbr[n * KNB + k]) * C;
        const float* w = W + k * C;
        for (int c8 = 0; c8 < C; c8 += 8) {
            union { uint4 q; bf16 h[8]; } u;
            u.q = *reinterpret_cast<const uint4*>(p + c8);
#pragma unroll
            for (int j = 0; j < 8; ++j) acc += (float)u.h[j] * w[c8 + j];
        }
    }
    logits[i] = acc;
}

// ---------------- softmax over N per (b,t) row ----------------
__global__ void softmax_k(const float* __restrict__ logits, float* __restrict__ out, int N) {
    __shared__ float red[256];
    int bt = blockIdx.x;
    const float* in = logits + (size_t)bt * N;
    float* o = out + (size_t)bt * N;
    float m = -3.4e38f;
    for (int i = threadIdx.x; i < N; i += blockDim.x) m = fmaxf(m, in[i]);
    red[threadIdx.x] = m; __syncthreads();
    for (int s = 128; s > 0; s >>= 1) {
        if (threadIdx.x < s) red[threadIdx.x] = fmaxf(red[threadIdx.x], red[threadIdx.x + s]);
        __syncthreads();
    }
    m = red[0]; __syncthreads();
    float sum = 0.f;
    for (int i = threadIdx.x; i < N; i += blockDim.x) sum += expf(in[i] - m);
    red[threadIdx.x] = sum; __syncthreads();
    for (int s = 128; s > 0; s >>= 1) {
        if (threadIdx.x < s) red[threadIdx.x] += red[threadIdx.x + s];
        __syncthreads();
    }
    float inv = 1.f / red[0];
    for (int i = threadIdx.x; i < N; i += blockDim.x) o[i] = expf(in[i] - m) * inv;
}

// ---------------- host side ----------------
static inline int cdiv(long long a, long long b) { return (int)((a + b - 1) / b); }

template<int CIN, int COUT>
static void launch_conv(const bf16* xin, const int* nbr, const unsigned int* pw,
                        const float* bias, bf16* out, int N, int Nin,
                        hipStream_t stream) {
    int tiles32 = (N + 31) / 32;
    long long blocks = (long long)BT * (COUT / 64) * ((tiles32 + 7) / 8);
    conv_wmma<CIN, COUT><<<(int)blocks, 256, 0, stream>>>(
        xin, nbr, (const uint4*)pw, bias, out, N, Nin, 1);
}

extern "C" void kernel_launch(void* const* d_in, const int* in_sizes, int n_in,
                              void* d_out, int out_size, void* d_ws, size_t ws_size,
                              hipStream_t stream) {
    const float* x    = (const float*)d_in[0];
    const int* nbr5   = (const int*)d_in[3];
    const int* nbr4   = (const int*)d_in[4];
    const int* nbr3   = (const int*)d_in[5];
    const int* pool54 = (const int*)d_in[6];
    const int* pool43 = (const int*)d_in[7];
    const int* up34   = (const int*)d_in[8];
    const int* up45   = (const int*)d_in[9];
    const float* W0 = (const float*)d_in[10]; const float* b0 = (const float*)d_in[11];
    const float* W1 = (const float*)d_in[12]; const float* b1 = (const float*)d_in[13];
    const float* W2 = (const float*)d_in[14]; const float* b2 = (const float*)d_in[15];
    const float* W7 = (const float*)d_in[16]; const float* b7 = (const float*)d_in[17];
    const float* W8 = (const float*)d_in[18]; const float* b8 = (const float*)d_in[19];
    const float* W9 = (const float*)d_in[20]; const float* b9 = (const float*)d_in[21];

    // workspace carve (256B aligned)
    char* p = (char*)d_ws;
    auto alloc = [&](size_t bytes) -> char* {
        char* r = p; p += (bytes + 255) & ~(size_t)255; return r;
    };
    bf16* xb   = (bf16*)alloc((size_t)BT * CN5 * 64  * 2);
    bf16* c0b  = (bf16*)alloc((size_t)BT * CN5 * 64  * 2);
    bf16* p0   = (bf16*)alloc((size_t)BT * CN4 * 64  * 2);
    bf16* c1   = (bf16*)alloc((size_t)BT * CN4 * 128 * 2);
    bf16* p1   = (bf16*)alloc((size_t)BT * CN3 * 128 * 2);
    bf16* c2   = (bf16*)alloc((size_t)BT * CN3 * 256 * 2);
    bf16* cat7 = (bf16*)alloc((size_t)BT * CN4 * 384 * 2);
    bf16* c7   = (bf16*)alloc((size_t)BT * CN4 * 128 * 2);
    bf16* cat8 = (bf16*)alloc((size_t)BT * CN5 * 192 * 2);
    bf16* c8   = (bf16*)alloc((size_t)BT * CN5 * 64  * 2);
    float* logits = (float*)alloc((size_t)BT * CN5 * 4);
    unsigned int* pw0 = (unsigned int*)alloc((size_t)KNB * 64  * 64  * 2);
    unsigned int* pw1 = (unsigned int*)alloc((size_t)KNB * 64  * 128 * 2);
    unsigned int* pw2 = (unsigned int*)alloc((size_t)KNB * 128 * 256 * 2);
    unsigned int* pw7 = (unsigned int*)alloc((size_t)KNB * 384 * 128 * 2);
    unsigned int* pw8 = (unsigned int*)alloc((size_t)KNB * 192 * 64  * 2);

    // convert input and pack weights to bf16 fragments
    int nx = BT * CN5 * 64;
    cvt_f32_bf16<<<cdiv(nx, 256), 256, 0, stream>>>(x, xb, nx);
    pack_w<<<cdiv((long long)KNB * 64  * 64  / 2, 256), 256, 0, stream>>>(W0, pw0, 64,  64);
    pack_w<<<cdiv((long long)KNB * 64  * 128 / 2, 256), 256, 0, stream>>>(W1, pw1, 64,  128);
    pack_w<<<cdiv((long long)KNB * 128 * 256 / 2, 256), 256, 0, stream>>>(W2, pw2, 128, 256);
    pack_w<<<cdiv((long long)KNB * 384 * 128 / 2, 256), 256, 0, stream>>>(W7, pw7, 384, 128);
    pack_w<<<cdiv((long long)KNB * 192 * 64  / 2, 256), 256, 0, stream>>>(W8, pw8, 192, 64);

    // U-Net forward
    launch_conv<64, 64>  (xb,   nbr5, pw0, b0, c0b, CN5, CN5, stream);   // conv0
    pool_max<<<cdiv((long long)BT * CN4 * 64, 256), 256, 0, stream>>>(c0b, pool54, p0, CN4, CN5, 64);
    launch_conv<64, 128> (p0,   nbr4, pw1, b1, c1,  CN4, CN4, stream);   // conv1
    pool_max<<<cdiv((long long)BT * CN3 * 128, 256), 256, 0, stream>>>(c1, pool43, p1, CN3, CN4, 128);
    launch_conv<128, 256>(p1,   nbr3, pw2, b2, c2,  CN3, CN3, stream);   // conv2
    concat_up<<<cdiv((long long)BT * CN4 * 384, 256), 256, 0, stream>>>(c2, up34, c1, cat7, CN4, CN3, 256, 128);
    launch_conv<384, 128>(cat7, nbr4, pw7, b7, c7,  CN4, CN4, stream);   // conv7
    concat_up<<<cdiv((long long)BT * CN5 * 192, 256), 256, 0, stream>>>(c7, up45, c0b, cat8, CN5, CN4, 128, 64);
    launch_conv<192, 64> (cat8, nbr5, pw8, b8, c8,  CN5, CN5, stream);   // conv8
    conv9_k<<<cdiv((long long)BT * CN5, 256), 256, 0, stream>>>(c8, nbr5, W9, b9, logits, CN5, 64);
    softmax_k<<<BT, 256, 0, stream>>>(logits, (float*)d_out, CN5);
}
